// NeuralHSMM_30133490549286
// MI455X (gfx1250) — matrix operand order repaired and checked by reading, not compiled, not run on hardware
//
#include <hip/hip_runtime.h>
#include <hip/hip_bf16.h>
#include <math.h>

typedef __attribute__((ext_vector_type(16))) _Float16 v16h;
typedef __attribute__((ext_vector_type(8)))  _Float16 v8h;
typedef __attribute__((ext_vector_type(8)))  float    v8f;

#define NEGV  (-1e30f)
#define LOGPI (-4.1588830833596715f)   /* -log(64) */

// ---- WMMA fragment maps (CDNA5 ISA 7.12.2, wave32) ----
// A (f16 16x32 MxK): lane row = lane&15; K = {k0..k0+7, k0+16..k0+23}, k0 = 8*(lane>=16).
// B (f16 32x16 KxN): lane n = lane&15; K = {k0..k0+7, k0+16..k0+23}, k0 = 8*(lane>=16) (same as A).
// C/D (f32 16x16):   lane n = lane&15; vgpr v -> m = v + 8*(lane>=16).

// Build a v16h from two contiguous 8-half (16B) LDS runs -> 2x ds_load_b128.
__device__ __forceinline__ v16h frag2x8(const _Float16* p0, const _Float16* p1) {
  v8h lo = *reinterpret_cast<const v8h*>(p0);
  v8h hi = *reinterpret_cast<const v8h*>(p1);
  return __builtin_shufflevector(lo, hi, 0, 1, 2, 3, 4, 5, 6, 7,
                                 8, 9, 10, 11, 12, 13, 14, 15);
}

// ============================================================
// 1) Wave-per-row GEMV (+ optional tanh) for the context MLPs.
// ============================================================
__global__ void gemv_act(const float* __restrict__ W, const float* __restrict__ v,
                         const float* __restrict__ bias, float* __restrict__ out,
                         int N, int K, int act) {
  int wave = threadIdx.x >> 5;
  int lane = threadIdx.x & 31;
  int row  = blockIdx.x * 8 + wave;
  if (row >= N) return;
  const float* wr = W + (size_t)row * K;
  float acc = 0.f;
  for (int j = lane; j < K; j += 32) acc += wr[j] * v[j];
#pragma unroll
  for (int o = 16; o > 0; o >>= 1) acc += __shfl_xor(acc, o, 32);
  if (lane == 0) {
    float r = acc + bias[row];
    if (act) r = tanhf(r);
    out[row] = r;
  }
}

// ============================================================
// 2) Prep: A softmax -> AhT[s'][s] (f16, TRANSPOSED for B-frags),
//    Dur softmax -> logDurT[d][s], Gaussian -> WaugT[s][0..127]
//    (row s: [-0.5*inv | mu*inv], contiguous in k), halfconst[s].
// ============================================================
__global__ void prep_kernel(const float* __restrict__ A_logits, const float* __restrict__ D_logits,
                            const float* __restrict__ mu, const float* __restrict__ log_var,
                            const float* __restrict__ dA, const float* __restrict__ dD,
                            const float* __restrict__ dE,
                            _Float16* __restrict__ AhT, float* __restrict__ logDurT,
                            _Float16* __restrict__ WaugT, float* __restrict__ halfconst) {
  int s = threadIdx.x;  // 0..63
  // --- A row softmax (TEMP = 1); store transposed ---
  float rowv[64];
  float mx = NEGV;
  for (int j = 0; j < 64; ++j) {
    float l = A_logits[s * 64 + j] + 0.1f * tanhf(dA[s * 64 + j]);
    rowv[j] = l; mx = fmaxf(mx, l);
  }
  float sum = 0.f;
  for (int j = 0; j < 64; ++j) { rowv[j] = __expf(rowv[j] - mx); sum += rowv[j]; }
  float isum = 1.f / sum;
  for (int j = 0; j < 64; ++j) AhT[j * 64 + s] = (_Float16)(rowv[j] * isum);
  // --- Dur row softmax -> logDur transposed [d][s] ---
  float dv[32];
  mx = NEGV;
  for (int d = 0; d < 32; ++d) {
    float l = D_logits[s * 32 + d] + 0.1f * tanhf(dD[s * 32 + d]);
    dv[d] = l; mx = fmaxf(mx, l);
  }
  sum = 0.f;
  for (int d = 0; d < 32; ++d) sum += __expf(dv[d] - mx);
  float lse = mx + __logf(sum);
  for (int d = 0; d < 32; ++d) logDurT[d * 64 + s] = dv[d] - lse;
  // --- Gaussian: augmented weights, row-contiguous in k ---
  float cst = 0.f;
  for (int f = 0; f < 64; ++f) {
    float mc = mu[s * 64 + f] + 0.1f * tanhf(dE[s * 64 + f]);
    float lv = log_var[s * 64 + f];
    float sp = (lv > 20.f) ? lv : log1pf(__expf(lv));  // softplus
    float var = sp + 0.001f;                            // MIN_COVAR
    float iv = 1.f / var;
    WaugT[s * 128 + f]      = (_Float16)(-0.5f * iv);   // pairs with x^2
    WaugT[s * 128 + 64 + f] = (_Float16)(mc * iv);      // pairs with x
    cst += mc * mc * iv + __logf(var);
  }
  halfconst[s] = 0.5f * (cst + 64.f * 1.8378770664093453f);  // + F*log(2*pi)
}

// ============================================================
// 3) logB GEMM via WMMA: (32768 x 128) @ (128 x 64).
//    xs2 = [x^2 | x] prebuilt in f16 LDS -> all frags are b128 loads.
// ============================================================
__global__ void logb_gemm(const float* __restrict__ x, const _Float16* __restrict__ WaugT,
                          const float* __restrict__ halfconst, float* __restrict__ logB) {
  __shared__ alignas(16) _Float16 xs2[32][128];   // [row][k]: k<64 -> x^2, k>=64 -> x
  __shared__ alignas(16) _Float16 WsT[64][128];   // [n][k]
  __shared__ float hc[64];
  int tid = threadIdx.x;
  int base = blockIdx.x * 32;
  for (int i = tid; i < 32 * 64; i += 256) {
    int row = i >> 6, f = i & 63;
    float xv = x[(size_t)base * 64 + i];
    xs2[row][f]      = (_Float16)(xv * xv);
    xs2[row][64 + f] = (_Float16)xv;
  }
  for (int i = tid; i < 64 * 128; i += 256) WsT[i >> 7][i & 127] = WaugT[i];
  if (tid < 64) hc[tid] = halfconst[tid];
  __syncthreads();

  int wave = tid >> 5, lane = tid & 31;
  int mt = wave >> 2, nt = wave & 3;
  int row = mt * 16 + (lane & 15);
  int nn  = nt * 16 + (lane & 15);
  int ka0 = (lane >> 4) << 3;    // A-frag K base offset
  int kb0 = (lane >> 4) << 3;    // B-frag K base offset
  v8f acc = {};
#pragma unroll
  for (int kc = 0; kc < 4; ++kc) {
    v16h a = frag2x8(&xs2[row][kc * 32 + ka0], &xs2[row][kc * 32 + ka0 + 16]);
    v16h b = frag2x8(&WsT[nn][kc * 32 + kb0], &WsT[nn][kc * 32 + kb0 + 16]);
    acc = __builtin_amdgcn_wmma_f32_16x16x32_f16(false, a, false, b,
                                                 (short)0, acc, false, false);
  }
  int hi = (lane >> 4) << 3;
#pragma unroll
  for (int v = 0; v < 8; ++v) {
    int m = base + mt * 16 + v + hi;
    logB[(size_t)m * 64 + nn] = acc[v] - hc[nn];
  }
}

// ============================================================
// 4) Cumsum over t per (b,s): Ccum[b][0..T][s]; coalesced in s.
// ============================================================
__global__ void cumsum_kernel(const float* __restrict__ logB, float* __restrict__ Ccum) {
  int b = blockIdx.x, s = threadIdx.x;
  float*       cb = Ccum + (size_t)b * 1025 * 64;
  const float* lb = logB + (size_t)b * 1024 * 64;
  float c = 0.f;
  cb[s] = 0.f;
  for (int t = 0; t < 1024; ++t) {
    c += lb[t * 64 + s];
    cb[(t + 1) * 64 + s] = c;
  }
}

// ============================================================
// 5) Persistent HSMM scan: one block per batch.
//    Per step: 4x ds_load_b128 + 2x v_wmma (A-matrix B-frags live
//    in VGPRs for the whole scan).  Ccum rows stream into the LDS
//    ring one step ahead via GLOBAL_LOAD_ASYNC_TO_LDS (ASYNCcnt),
//    hiding the global load latency in the serial chain.
// ============================================================
__global__ void hsmm_scan(const float* __restrict__ Ccum, const _Float16* __restrict__ AhT,
                          const float* __restrict__ logDurT, float* __restrict__ out) {
  __shared__ alignas(16) float    bufL[32][64];    // ring: slot j&31 holds alpha_j
  __shared__ alignas(16) _Float16 Pl[32][64];      // exp(buf - rowmax)
  __shared__ alignas(16) float    mrow[32];
  __shared__ alignas(16) float    transL[32][64];
  __shared__ alignas(16) float    ringC[34][64];   // Ccum rows t-32 .. t+1
  __shared__ alignas(16) _Float16 AhTL[64][64];    // A transposed: [s'][s]
  __shared__ alignas(16) float    logDurL[32][64];
  __shared__ float pM[4][64], pS[4][64];
  int tid = threadIdx.x;
  int b = blockIdx.x;
  const float* cb = Ccum + (size_t)b * 1025 * 64;

  for (int i = tid; i < 64 * 64; i += 256) AhTL[i >> 6][i & 63] = AhT[i];
  for (int i = tid; i < 32 * 64; i += 256) {
    logDurL[i >> 6][i & 63] = logDurT[i];
    bufL[i >> 6][i & 63] = NEGV;
  }
  if (tid < 128) ringC[tid >> 6][tid & 63] = cb[tid];  // rows 0,1 (sync)
  __syncthreads();

  int wave = tid >> 5, lane = tid & 31;
  int mt = wave >> 2, nt = wave & 3;
  int arow = mt * 16 + (lane & 15);
  int nn   = nt * 16 + (lane & 15);
  int ka0  = (lane >> 4) << 3;
  int kb0  = (lane >> 4) << 3;
  int hi   = (lane >> 4) << 3;
  // constant transition-matrix B-fragments, resident in VGPRs all scan long
  v16h bf0 = frag2x8(&AhTL[nn][kb0],      &AhTL[nn][kb0 + 16]);
  v16h bf1 = frag2x8(&AhTL[nn][kb0 + 32], &AhTL[nn][kb0 + 48]);

  for (int t = 0; t < 1024; ++t) {
    // (a) wave 0: wait for last step's async row (t+1), then async-
    //     prefetch row t+2 into ring slot (t+2)%34 (holds row t-32,
    //     which is no longer read at step t -> race-free overwrite).
    if (wave == 0) {
      asm volatile("s_wait_asynccnt 0x0" ::: "memory");
      int tp2 = t + 2;
      if (tp2 <= 1024) {
        unsigned lds_dst = (unsigned)(uintptr_t)&ringC[tp2 % 34][lane * 2];
        unsigned long long gsrc =
            (unsigned long long)(uintptr_t)(cb + (size_t)tp2 * 64 + lane * 2);
        asm volatile("global_load_async_to_lds_b64 %0, %1, off"
                     :: "v"(lds_dst), "v"(gsrc) : "memory");
      }
    }
    // (b) row maxima + P = exp(buf - m), 8 threads per d-row
    {
      int d = tid >> 3, j0 = (tid & 7) * 8;
      int p = (t + 31 - d) & 31;  // ring slot of alpha_{t-1-d}
      float mx = NEGV;
#pragma unroll
      for (int j = 0; j < 8; ++j) mx = fmaxf(mx, bufL[p][j0 + j]);
#pragma unroll
      for (int o = 1; o < 8; o <<= 1) mx = fmaxf(mx, __shfl_xor(mx, o, 8));
      if ((tid & 7) == 0) mrow[d] = mx;
#pragma unroll
      for (int j = 0; j < 8; ++j) Pl[d][j0 + j] = (_Float16)__expf(bufL[p][j0 + j] - mx);
    }
    __syncthreads();
    // (c) trans via WMMA (exp-domain): 4x ds_load_b128 + 2x v_wmma
    {
      v16h a0 = frag2x8(&Pl[arow][ka0],      &Pl[arow][ka0 + 16]);
      v8f acc = {};
      acc = __builtin_amdgcn_wmma_f32_16x16x32_f16(false, a0, false, bf0,
                                                   (short)0, acc, false, false);
      v16h a1 = frag2x8(&Pl[arow][ka0 + 32], &Pl[arow][ka0 + 48]);
      acc = __builtin_amdgcn_wmma_f32_16x16x32_f16(false, a1, false, bf1,
                                                   (short)0, acc, false, false);
#pragma unroll
      for (int v = 0; v < 8; ++v) {
        int m = mt * 16 + v + hi;
        float sv = acc[v];
        transL[m][nn] = mrow[m] + (sv > 0.f ? __logf(sv) : NEGV);
      }
    }
    __syncthreads();
    // (d1) partial alpha logsumexp over 8 d's per group (4 groups x 64 s')
    {
      int sp = tid & 63, g = tid >> 6;
      float M = NEGV, S = 0.f;
      float ct1 = ringC[(t + 1) % 34][sp];
      int d0 = g * 8;
#pragma unroll
      for (int dd = 0; dd < 8; ++dd) {
        int d = d0 + dd, di = d + 1;
        float pv;
        if (di == t + 1)  pv = LOGPI;
        else if (di <= t) pv = transL[d][sp];
        else continue;
        float val = pv + logDurL[d][sp] + ct1 - ringC[(t + 1 - di) % 34][sp];
        if (val > M) { S = S * __expf(M - val) + 1.f; M = val; }
        else         { S += __expf(val - M); }
      }
      pM[g][sp] = M; pS[g][sp] = S;
    }
    __syncthreads();
    // (d2) merge the 4 partials, write alpha_t into the ring
    if (tid < 64) {
      int sp = tid;
      float M = pM[0][sp], S = pS[0][sp];
#pragma unroll
      for (int g = 1; g < 4; ++g) {
        float m2 = pM[g][sp], s2 = pS[g][sp];
        if (m2 > M) { S = S * __expf(M - m2) + s2; M = m2; }
        else        { S += s2 * __expf(m2 - M); }
      }
      bufL[t & 31][sp] = (S > 0.f) ? (M + __logf(S)) : NEGV;
    }
    __syncthreads();
  }
  // drain any outstanding async op before exit
  if (wave == 0) asm volatile("s_wait_asynccnt 0x0" ::: "memory");
  // final: out[b] = logsumexp_s alpha_{T-1}
  if (tid == 0) {
    float M = NEGV, S = 0.f;
    for (int s = 0; s < 64; ++s) {
      float v = bufL[1023 & 31][s];
      if (v > M) { S = S * __expf(M - v) + 1.f; M = v; }
      else       { S += __expf(v - M); }
    }
    out[b] = M + __logf(S);
  }
}

// ============================================================
extern "C" void kernel_launch(void* const* d_in, const int* in_sizes, int n_in,
                              void* d_out, int out_size, void* d_ws, size_t ws_size,
                              hipStream_t stream) {
  const float* x        = (const float*)d_in[0];
  const float* context  = (const float*)d_in[1];
  const float* A_logits = (const float*)d_in[2];
  const float* D_logits = (const float*)d_in[3];
  const float* mu       = (const float*)d_in[4];
  const float* log_var  = (const float*)d_in[5];
  const float* tA_w1 = (const float*)d_in[6];
  const float* tA_b1 = (const float*)d_in[7];
  const float* tA_w2 = (const float*)d_in[8];
  const float* tA_b2 = (const float*)d_in[9];
  const float* tD_w1 = (const float*)d_in[10];
  const float* tD_b1 = (const float*)d_in[11];
  const float* tD_w2 = (const float*)d_in[12];
  const float* tD_b2 = (const float*)d_in[13];
  const float* tE_w1 = (const float*)d_in[14];
  const float* tE_b1 = (const float*)d_in[15];
  const float* tE_w2 = (const float*)d_in[16];
  const float* tE_b2 = (const float*)d_in[17];
  float* out = (float*)d_out;
  (void)in_sizes; (void)n_in; (void)out_size; (void)ws_size;

  char* ws = (char*)d_ws;
  size_t off = 0;
  auto alloc = [&](size_t bytes) -> void* {
    void* p = ws + off;
    off = (off + bytes + 255) & ~(size_t)255;
    return p;
  };
  float*    hbuf      = (float*)alloc(4096 * 4);
  float*    dA        = (float*)alloc(4096 * 4);
  float*    dD        = (float*)alloc(2048 * 4);
  float*    dE        = (float*)alloc(4096 * 4);
  _Float16* AhT       = (_Float16*)alloc(4096 * 2);
  float*    logDurT   = (float*)alloc(2048 * 4);
  _Float16* WaugT     = (_Float16*)alloc(8192 * 2);
  float*    halfconst = (float*)alloc(64 * 4);
  float*    logB      = (float*)alloc((size_t)32768 * 64 * 4);
  float*    Ccum      = (float*)alloc((size_t)32 * 1025 * 64 * 4);

  // context MLPs (stream-serialized; hbuf reused)
  gemv_act<<<4096 / 8, 256, 0, stream>>>(tA_w1, context, tA_b1, hbuf, 4096, 128, 1);
  gemv_act<<<4096 / 8, 256, 0, stream>>>(tA_w2, hbuf,    tA_b2, dA,   4096, 4096, 0);
  gemv_act<<<2048 / 8, 256, 0, stream>>>(tD_w1, context, tD_b1, hbuf, 2048, 128, 1);
  gemv_act<<<2048 / 8, 256, 0, stream>>>(tD_w2, hbuf,    tD_b2, dD,   2048, 2048, 0);
  gemv_act<<<4096 / 8, 256, 0, stream>>>(tE_w1, context, tE_b1, hbuf, 4096, 128, 1);
  gemv_act<<<4096 / 8, 256, 0, stream>>>(tE_w2, hbuf,    tE_b2, dE,   4096, 4096, 0);
  // parameter prep
  prep_kernel<<<1, 64, 0, stream>>>(A_logits, D_logits, mu, log_var, dA, dD, dE,
                                    AhT, logDurT, WaugT, halfconst);
  // emission log-likelihoods via WMMA GEMM
  logb_gemm<<<1024, 256, 0, stream>>>(x, WaugT, halfconst, logB);
  // cumulative sums over time
  cumsum_kernel<<<32, 64, 0, stream>>>(logB, Ccum);
  // persistent per-batch HSMM forward scan (WMMA + async Ccum stream)
  hsmm_scan<<<32, 256, 0, stream>>>(Ccum, AhT, logDurT, out);
}
